// InfoNCELoss_22909355557339
// MI455X (gfx1250) — matrix operand order repaired and checked by reading, not compiled, not run on hardware
//
#include <hip/hip_runtime.h>
#include <hip/hip_bf16.h>

#define USE_ASYNC_LDS 1   // set to 0 to fall back to global->reg->ds_store staging

typedef _Float16 v16h __attribute__((ext_vector_type(16)));
typedef _Float16 v8h  __attribute__((ext_vector_type(8)));
typedef _Float16 v4h  __attribute__((ext_vector_type(4)));
typedef float    v8f  __attribute__((ext_vector_type(8)));
typedef float    v4f  __attribute__((ext_vector_type(4)));

#define DIM     512
#define BANCH   4096
#define NPOS    4096
#define NNEG    16384
#define TEMP_INV 14.2857142857142857f   // 1/0.07
#define EPS_     1e-8f

// Block GEMM tile
#define BM 128
#define BN 256
#define BK 32
#define KSTEPS (DIM / BK)   // 16

// ---------------------------------------------------------------------------
// Row L2-normalize fp32 -> f16 (one wave32 per row, 8 rows per 256-thread block)
// ---------------------------------------------------------------------------
__global__ void __launch_bounds__(256)
normalize_f16_kernel(const float* __restrict__ in, _Float16* __restrict__ out, int rows) {
    const int wave = threadIdx.x >> 5;
    const int lane = threadIdx.x & 31;
    const int row  = blockIdx.x * (blockDim.x >> 5) + wave;
    if (row >= rows) return;

    const float* src = in + (size_t)row * DIM;
    float vals[16];
    float ss = 0.f;
#pragma unroll
    for (int j = 0; j < 4; ++j) {
        v4f v = *(const v4f*)(src + lane * 4 + j * 128);
#pragma unroll
        for (int c = 0; c < 4; ++c) { vals[j * 4 + c] = v[c]; ss += v[c] * v[c]; }
    }
#pragma unroll
    for (int off = 1; off < 32; off <<= 1) ss += __shfl_xor(ss, off, 32);
    const float inv = 1.0f / fmaxf(sqrtf(ss), 1e-12f);

    _Float16* dst = out + (size_t)row * DIM;
#pragma unroll
    for (int j = 0; j < 4; ++j) {
        v4h h;
#pragma unroll
        for (int c = 0; c < 4; ++c) h[c] = (_Float16)(vals[j * 4 + c] * inv);
        *(v4h*)(dst + lane * 4 + j * 128) = h;
    }
}

__global__ void zero_kernel(float* __restrict__ p, int n) {
    int i = blockIdx.x * blockDim.x + threadIdx.x;
    if (i < n) p[i] = 0.f;
}

// ---------------------------------------------------------------------------
// CDNA5 async global->LDS copy (ASYNCcnt-tracked), ISA §15.18.3 op 98.
//  vdst = LDS byte offset (VGPR), vaddr = 64-bit global address (VGPR pair)
// ---------------------------------------------------------------------------
__device__ __forceinline__ void async_cp_b128(unsigned lds_off, const void* gaddr) {
    asm volatile("global_load_async_to_lds_b128 %0, %1, off"
                 :: "v"(lds_off), "v"(gaddr) : "memory");
}
__device__ __forceinline__ void wait_asynccnt0() {
    asm volatile("s_wait_asynccnt 0x0" ::: "memory");
}
__device__ __forceinline__ unsigned lds_off_u32(const void* p) {
    // generic pointer to __shared__: addr[31:0] is the LDS byte offset (aperture layout)
    return (unsigned)(unsigned long long)p;
}

// ---------------------------------------------------------------------------
// WMMA fragment helper (gfx1250 wave32 layout, 16x16x32 f16)
// ---------------------------------------------------------------------------
__device__ __forceinline__ v16h frag_cat(const _Float16* p0, const _Float16* p1) {
    v8h lo = *(const v8h*)p0;   // b128
    v8h hi = *(const v8h*)p1;   // b128
    v16h r;
#pragma unroll
    for (int i = 0; i < 8; ++i) { r[i] = lo[i]; r[i + 8] = hi[i]; }
    return r;
}

// ---------------------------------------------------------------------------
// 128x256 block tile, 8 waves (wave tile 64x64 = 16 acc tiles), double-buffered
// LDS pipeline with async global->LDS staging.
//  sA = [2][BM*BK] f16 (8 KB/buf), sB = [2][BN*BK] f16 (16 KB/buf) -> 48 KB LDS
// All 8 fragments (4 A + 4 B) are preloaded into distinct register sets before
// the 16-WMMA burst so the compiler can use partial s_wait_dscnt instead of
// full drains between WMMA quartets.
// ---------------------------------------------------------------------------
__device__ __forceinline__ void block_gemm(const _Float16* __restrict__ Ag,
                                           const _Float16* __restrict__ Bg,
                                           _Float16* __restrict__ sA,
                                           _Float16* __restrict__ sB,
                                           v8f acc[16]) {
    const int tid  = threadIdx.x;
    const int lane = tid & 31;
    const int wave = tid >> 5;
    const int half = lane >> 4;
    const int lm   = lane & 15;
    const int wm   = wave >> 2;   // 0..1 -> M offset wm*64
    const int wn   = wave & 3;    // 0..3 -> N offset wn*64

    // per-thread staging chunks: slice row = tid>>2, 8-f16 sub-chunk = tid&3
    const int rA = tid >> 2;
    const int kA = (tid & 3) * 8;

#if USE_ASYNC_LDS
    const unsigned sA_off = lds_off_u32(sA);
    const unsigned sB_off = lds_off_u32(sB);
    auto stage = [&](int kb, int buf) {
        const unsigned la = sA_off + (unsigned)buf * (BM * BK * 2) + (unsigned)(rA * BK + kA) * 2;
        const unsigned lb = sB_off + (unsigned)buf * (BN * BK * 2) + (unsigned)(rA * BK + kA) * 2;
        async_cp_b128(la,                 Ag + (size_t)(rA      ) * DIM + kb + kA);
        async_cp_b128(la + 64 * BK * 2,   Ag + (size_t)(rA +  64) * DIM + kb + kA);
#pragma unroll
        for (int j = 0; j < 4; ++j)
            async_cp_b128(lb + j * 64 * BK * 2, Bg + (size_t)(rA + 64 * j) * DIM + kb + kA);
    };
    stage(0, 0);
    wait_asynccnt0();
    __syncthreads();
#else
    v8h st[6];
    auto stage_load = [&](int kb) {
        st[0] = *(const v8h*)(Ag + (size_t)(rA      ) * DIM + kb + kA);
        st[1] = *(const v8h*)(Ag + (size_t)(rA +  64) * DIM + kb + kA);
#pragma unroll
        for (int j = 0; j < 4; ++j)
            st[2 + j] = *(const v8h*)(Bg + (size_t)(rA + 64 * j) * DIM + kb + kA);
    };
    auto stage_store = [&](int buf) {
        _Float16* a = sA + buf * (BM * BK);
        _Float16* b = sB + buf * (BN * BK);
        *(v8h*)(a + (rA      ) * BK + kA) = st[0];
        *(v8h*)(a + (rA +  64) * BK + kA) = st[1];
#pragma unroll
        for (int j = 0; j < 4; ++j)
            *(v8h*)(b + (rA + 64 * j) * BK + kA) = st[2 + j];
    };
    stage_load(0);
    stage_store(0);
    __syncthreads();
#endif

    for (int ks = 0; ks < KSTEPS; ++ks) {
        const int buf = ks & 1;
        const bool more = (ks + 1 < KSTEPS);

#if USE_ASYNC_LDS
        if (more) stage((ks + 1) * BK, buf ^ 1);  // DMA into other buffer, overlaps compute
#else
        if (more) stage_load((ks + 1) * BK);
#endif

        const _Float16* a_lds = sA + buf * (BM * BK);
        const _Float16* b_lds = sB + buf * (BN * BK);

        // preload all fragments into distinct register sets (enables partial waits)
        v16h a[4], b[4];
#pragma unroll
        for (int mt = 0; mt < 4; ++mt) {
            const _Float16* pa = a_lds + (wm * 64 + mt * 16 + lm) * BK + half * 8;
            a[mt] = frag_cat(pa, pa + 16);
        }
#pragma unroll
        for (int nt = 0; nt < 4; ++nt) {
            const _Float16* pb = b_lds + (wn * 64 + nt * 16 + lm) * BK + half * 8;
            b[nt] = frag_cat(pb, pb + 16);
        }
#pragma unroll
        for (int nt = 0; nt < 4; ++nt)
#pragma unroll
            for (int mt = 0; mt < 4; ++mt)
                acc[mt * 4 + nt] = __builtin_amdgcn_wmma_f32_16x16x32_f16(
                    false, a[mt], false, b[nt], (short)0, acc[mt * 4 + nt], false, false);

        if (more) {
#if USE_ASYNC_LDS
            wait_asynccnt0();       // our 6 copies into buf^1 are done
#else
            stage_store(buf ^ 1);
#endif
            __syncthreads();        // publish buf^1 to all waves
        }
    }
}

// element (acc[mt*4+nt], VGPR v, lane) -> row M = wm*64 + mt*16 + v + 8*half
//                                         col N = wn*64 + nt*16 + lm

// ---------------------------------------------------------------------------
// Negatives: per-anchor sum_k exp(sim/T) + global sum(sim/T)
// grid = (BANCH/BM, NNEG/BN), block = 256
// ---------------------------------------------------------------------------
__global__ void __launch_bounds__(256)
neg_pass_kernel(const _Float16* __restrict__ A16, const _Float16* __restrict__ N16,
                float* __restrict__ negExpSum, float* __restrict__ accum) {
    __shared__ _Float16 sA[2 * BM * BK];
    __shared__ _Float16 sB[2 * BN * BK];

    const int lane = threadIdx.x & 31;
    const int wave = threadIdx.x >> 5;
    const int half = lane >> 4, lm = lane & 15;
    const int wm = wave >> 2;
    const int mbase = blockIdx.x * BM;

    v8f acc[16];
#pragma unroll
    for (int t = 0; t < 16; ++t) acc[t] = 0.0f;

    block_gemm(A16 + (size_t)mbase * DIM, N16 + (size_t)(blockIdx.y * BN) * DIM, sA, sB, acc);

    float simSum = 0.f;
#pragma unroll
    for (int mt = 0; mt < 4; ++mt) {
#pragma unroll
        for (int v = 0; v < 8; ++v) {
            float e = 0.f;
#pragma unroll
            for (int nt = 0; nt < 4; ++nt) {
                float s = acc[mt * 4 + nt][v] * TEMP_INV;
                simSum += s;
                e += __expf(s);
            }
#pragma unroll
            for (int off = 1; off < 16; off <<= 1) e += __shfl_xor(e, off, 32); // within half
            if (lm == 0)
                atomicAdd(&negExpSum[mbase + wm * 64 + mt * 16 + half * 8 + v], e);
        }
    }
#pragma unroll
    for (int off = 1; off < 32; off <<= 1) simSum += __shfl_xor(simSum, off, 32);
    if (lane == 0) atomicAdd(&accum[0], simSum);
}

// ---------------------------------------------------------------------------
// Positives: sum(pos_sim/T) + sum(loss) using negExpSum
// grid = (BANCH/BM, NPOS/BN), block = 256
// ---------------------------------------------------------------------------
__global__ void __launch_bounds__(256)
pos_pass_kernel(const _Float16* __restrict__ A16, const _Float16* __restrict__ P16,
                const float* __restrict__ negExpSum, float* __restrict__ accum) {
    __shared__ _Float16 sA[2 * BM * BK];
    __shared__ _Float16 sB[2 * BN * BK];

    const int lane = threadIdx.x & 31;
    const int wave = threadIdx.x >> 5;
    const int half = lane >> 4, lm = lane & 15;
    const int wm = wave >> 2;
    const int mbase = blockIdx.x * BM;

    v8f acc[16];
#pragma unroll
    for (int t = 0; t < 16; ++t) acc[t] = 0.0f;

    block_gemm(A16 + (size_t)mbase * DIM, P16 + (size_t)(blockIdx.y * BN) * DIM, sA, sB, acc);

    float posSum = 0.f, lossSum = 0.f;
#pragma unroll
    for (int mt = 0; mt < 4; ++mt) {
#pragma unroll
        for (int v = 0; v < 8; ++v) {
            const float S = negExpSum[mbase + wm * 64 + mt * 16 + half * 8 + v];
#pragma unroll
            for (int nt = 0; nt < 4; ++nt) {
                float s  = acc[mt * 4 + nt][v] * TEMP_INV;
                posSum  += s;
                float pe = __expf(s);
                lossSum -= __logf(pe / (pe + S) + EPS_);
            }
        }
    }
#pragma unroll
    for (int off = 1; off < 32; off <<= 1) {
        posSum  += __shfl_xor(posSum, off, 32);
        lossSum += __shfl_xor(lossSum, off, 32);
    }
    if (lane == 0) {
        atomicAdd(&accum[1], posSum);
        atomicAdd(&accum[2], lossSum);
    }
}

__global__ void finalize_kernel(const float* __restrict__ accum, float* __restrict__ out) {
    const float invBP = 1.0f / ((float)BANCH * (float)NPOS);
    const float invBN = 1.0f / ((float)BANCH * (float)NNEG);
    float meanPos = accum[1] * invBP;
    float meanNeg = accum[0] * invBN;
    out[0] = accum[2] * invBP;     // total_loss
    out[1] = meanPos;
    out[2] = meanNeg;
    out[3] = meanPos - meanNeg;
}

// ---------------------------------------------------------------------------
extern "C" void kernel_launch(void* const* d_in, const int* in_sizes, int n_in,
                              void* d_out, int out_size, void* d_ws, size_t ws_size,
                              hipStream_t stream) {
    const float* A = (const float*)d_in[0];   // [4096,512]
    const float* P = (const float*)d_in[1];   // [4096,512]
    const float* N = (const float*)d_in[2];   // [16384,512]

    char* ws = (char*)d_ws;
    const size_t szA = (size_t)BANCH * DIM * sizeof(_Float16);   // 4 MiB
    const size_t szP = (size_t)NPOS  * DIM * sizeof(_Float16);   // 4 MiB
    const size_t szN = (size_t)NNEG  * DIM * sizeof(_Float16);   // 16 MiB
    _Float16* A16 = (_Float16*)ws;
    _Float16* P16 = (_Float16*)(ws + szA);
    _Float16* N16 = (_Float16*)(ws + szA + szP);
    float* negExpSum = (float*)(ws + szA + szP + szN);           // 4096 floats
    float* accum     = negExpSum + BANCH;                        // [negSum, posSum, lossSum]

    normalize_f16_kernel<<<BANCH / 8, 256, 0, stream>>>(A, A16, BANCH);
    normalize_f16_kernel<<<NPOS  / 8, 256, 0, stream>>>(P, P16, NPOS);
    normalize_f16_kernel<<<NNEG  / 8, 256, 0, stream>>>(N, N16, NNEG);

    zero_kernel<<<(BANCH + 3 + 255) / 256, 256, 0, stream>>>(negExpSum, BANCH + 3);

    neg_pass_kernel<<<dim3(BANCH / BM, NNEG / BN), 256, 0, stream>>>(A16, N16, negExpSum, accum);
    pos_pass_kernel<<<dim3(BANCH / BM, NPOS / BN), 256, 0, stream>>>(A16, P16, negExpSum, accum);

    finalize_kernel<<<1, 1, 0, stream>>>(accum, (float*)d_out);
}
